// SparrowSegHead_89197880803898
// MI455X (gfx1250) — compile-verified, matrix operand order checked by reading, not run on hardware
//
#include <hip/hip_runtime.h>
#include <hip/hip_bf16.h>

typedef __attribute__((ext_vector_type(16))) _Float16 v16h;
typedef __attribute__((ext_vector_type(8)))  _Float16 v8h;
typedef __attribute__((ext_vector_type(8)))  float    v8f;

#define PN   4800     // bs*Q
#define KK   4096     // H*W
#define TT   400      // targets
#define TP   448      // targets padded to 7*64
#define NC   81
#define HM   128
#define BM   64
#define BN   64
#define BK   32
#define LDT  40       // padded LDS row stride in f16 (32 + 8)
#define KSTEPS (KK / BK)   // 128
#define TILEB (BM * LDT * 2)   // bytes per LDS tile (5120)
#define EPSV 1e-5f

// ---------------------------------------------------------------- kernel 1
__global__ __launch_bounds__(256) void sg_softmax(const float* __restrict__ logits,
                                                  float* __restrict__ probs) {
    const int row  = blockIdx.x * 8 + (threadIdx.x >> 5);
    const int lane = threadIdx.x & 31;
    if (row >= PN) return;
    const float* in = logits + (size_t)row * NC;
    float x0 = (lane      < NC) ? in[lane]      : -1e30f;
    float x1 = (lane + 32 < NC) ? in[lane + 32] : -1e30f;
    float x2 = (lane + 64 < NC) ? in[lane + 64] : -1e30f;
    float m = fmaxf(x0, fmaxf(x1, x2));
    #pragma unroll
    for (int o = 16; o; o >>= 1) m = fmaxf(m, __shfl_xor(m, o, 32));
    float e0 = __expf(x0 - m);
    float e1 = (lane + 32 < NC) ? __expf(x1 - m) : 0.f;
    float e2 = (lane + 64 < NC) ? __expf(x2 - m) : 0.f;
    float s = e0 + e1 + e2;
    #pragma unroll
    for (int o = 16; o; o >>= 1) s += __shfl_xor(s, o, 32);
    const float inv = 1.0f / s;
    float* outp = probs + (size_t)row * NC;
    if (lane      < NC) outp[lane]      = e0 * inv;
    if (lane + 32 < NC) outp[lane + 32] = e1 * inv;
    if (lane + 64 < NC) outp[lane + 64] = e2 * inv;
}

// ---------------------------------------------------------------- kernel 2
__global__ __launch_bounds__(256) void sg_prep_pred(const float* __restrict__ pm,
                                                    _Float16* __restrict__ Amat,
                                                    float* __restrict__ sum_lognp,
                                                    float* __restrict__ sum_sig) {
    const int p = blockIdx.x;
    const float* row = pm + (size_t)p * KK;
    _Float16* a1 = Amat + (size_t)p * KK;
    _Float16* a2 = Amat + (size_t)(PN + p) * KK;
    float sln = 0.f, ssg = 0.f;
    for (int j = threadIdx.x; j < KK; j += 256) {
        const float m   = row[j];
        const float sig = 1.0f / (1.0f + __expf(-m));
        const float lnp = fminf(-m, 0.f) - log1pf(__expf(-fabsf(m)));  // log_sigmoid(-m)
        a1[j] = (_Float16)m;
        a2[j] = (_Float16)sig;
        sln += lnp;
        ssg += sig;
    }
    #pragma unroll
    for (int o = 16; o; o >>= 1) { sln += __shfl_xor(sln, o, 32); ssg += __shfl_xor(ssg, o, 32); }
    __shared__ float red[16];
    const int wid = threadIdx.x >> 5;
    if ((threadIdx.x & 31) == 0) { red[wid] = sln; red[8 + wid] = ssg; }
    __syncthreads();
    if (threadIdx.x == 0) {
        float a = 0.f, b = 0.f;
        #pragma unroll
        for (int i = 0; i < 8; ++i) { a += red[i]; b += red[8 + i]; }
        sum_lognp[p] = a;
        sum_sig[p]   = b;
    }
}

// ---------------------------------------------------------------- kernel 3
__global__ __launch_bounds__(256) void sg_prep_tgt(const float* __restrict__ tm,
                                                   _Float16* __restrict__ Bmat,
                                                   float* __restrict__ colsum) {
    const int t = blockIdx.x;
    _Float16* outr = Bmat + (size_t)t * KK;
    if (t >= TT) {
        for (int j = threadIdx.x; j < KK; j += 256) outr[j] = (_Float16)0.f;
        if (threadIdx.x == 0) colsum[t] = 0.f;
        return;
    }
    const float W4[4] = {0.125f, 0.375f, 0.375f, 0.125f};
    const float* src = tm + (size_t)t * (HM * HM);
    float s = 0.f;
    for (int j = threadIdx.x; j < KK; j += 256) {
        const int y = j >> 6, x = j & 63;
        float num_y = 0.f, den_y = 0.f;
        #pragma unroll
        for (int a = 0; a < 4; ++a) {
            const int ry = 2 * y - 1 + a;
            if (ry < 0 || ry >= HM) continue;
            float num_x = 0.f, den_x = 0.f;
            #pragma unroll
            for (int b = 0; b < 4; ++b) {
                const int rx = 2 * x - 1 + b;
                if (rx < 0 || rx >= HM) continue;
                num_x += W4[b] * src[ry * HM + rx];
                den_x += W4[b];
            }
            num_y += W4[a] * (num_x / den_x);
            den_y += W4[a];
        }
        const float v = num_y / den_y;
        outr[j] = (_Float16)v;
        s += v;
    }
    #pragma unroll
    for (int o = 16; o; o >>= 1) s += __shfl_xor(s, o, 32);
    __shared__ float red[8];
    if ((threadIdx.x & 31) == 0) red[threadIdx.x >> 5] = s;
    __syncthreads();
    if (threadIdx.x == 0) {
        float a = 0.f;
        #pragma unroll
        for (int i = 0; i < 8; ++i) a += red[i];
        colsum[t] = a;
    }
}

// ---------------------------------------------------------------- kernel 4
// dual WMMA GEMM with async global->LDS DMA (no VGPR staging) + fused epilogue.
__device__ __forceinline__ void sg_async16(unsigned lds_dst, unsigned goff,
                                           unsigned long long base) {
    // GVS mode: copies global saddr + vaddr into LDS at address vdst, 16 bytes per lane, ASYNCcnt
    asm volatile("global_load_async_to_lds_b128 %0, %1, %2"
                 :: "v"(lds_dst), "v"(goff), "s"(base) : "memory");
}

__device__ __forceinline__ void sg_mma_step(const _Float16* tA1, const _Float16* tA2,
                                            const _Float16* tB, int wm, int wn,
                                            int lrow, int ko, v8f acc[2][2][2]) {
    union { v16h v; v8h h[2]; } u;
    v16h bf[2];
    #pragma unroll
    for (int nt = 0; nt < 2; ++nt) {
        const int brow = (wn * 32 + nt * 16 + lrow) * LDT + ko;
        u.h[0] = *(const v8h*)&tB[brow];
        u.h[1] = *(const v8h*)&tB[brow + 16];
        bf[nt] = u.v;
    }
    #pragma unroll
    for (int mt = 0; mt < 2; ++mt) {
        const int arow = (wm * 32 + mt * 16 + lrow) * LDT + ko;
        u.h[0] = *(const v8h*)&tA1[arow];
        u.h[1] = *(const v8h*)&tA1[arow + 16];
        const v16h af1 = u.v;
        u.h[0] = *(const v8h*)&tA2[arow];
        u.h[1] = *(const v8h*)&tA2[arow + 16];
        const v16h af2 = u.v;
        #pragma unroll
        for (int nt = 0; nt < 2; ++nt) {
            acc[0][mt][nt] = __builtin_amdgcn_wmma_f32_16x16x32_f16(
                false, af1, false, bf[nt], (short)0, acc[0][mt][nt], false, false);
            acc[1][mt][nt] = __builtin_amdgcn_wmma_f32_16x16x32_f16(
                false, af2, false, bf[nt], (short)0, acc[1][mt][nt], false, false);
        }
    }
}

__global__ __launch_bounds__(128, 1) void sg_gemm_epilogue(
    const _Float16* __restrict__ Amat,    // [2*PN, KK]
    const _Float16* __restrict__ Bmat,    // [TP, KK]
    const float4*   __restrict__ pboxes,  // [PN]
    const float4*   __restrict__ tboxes,  // [TT]
    const float*    __restrict__ probs,   // [PN, NC]
    const float*    __restrict__ sum_lognp,
    const float*    __restrict__ sum_sig,
    const float*    __restrict__ colsum_t,
    const int*      __restrict__ labels,
    float*          __restrict__ out)     // [PN, TT]
{
    __shared__ __align__(16) _Float16 lds[4][3][BM * LDT];  // 4-deep ring, ~60 KB

    const int tid = threadIdx.x;
    const int bn0 = blockIdx.x * BN;   // 7 blocks over padded T
    const int bm0 = blockIdx.y * BM;   // 75 blocks over P

    // cooperative async-DMA mapping: each thread owns 16B chunks in rows r0, r0+32
    const int r0 = tid >> 2;           // 0..31
    const int c0 = (tid & 3) * 8;      // 0,8,16,24
    const unsigned offA = (unsigned)(((bm0 + r0) * KK + c0) * 2);
    const unsigned offB = (unsigned)(((bn0 + r0) * KK + c0) * 2);
    const unsigned ROW1 = 32u * KK * 2u;          // +32 rows in global (262144 B)
    const unsigned long long baseA1 = (unsigned long long)(uintptr_t)Amat;
    const unsigned long long baseA2 = (unsigned long long)(uintptr_t)(Amat + (size_t)PN * KK);
    const unsigned long long baseB  = (unsigned long long)(uintptr_t)Bmat;
    const unsigned ldsb = (unsigned)(uintptr_t)(&lds[0][0][0]);
    const unsigned thr  = (unsigned)((r0 * LDT + c0) * 2);

#define SG_DST(buf, tile_) (ldsb + (unsigned)(((buf) * 3 + (tile_)) * TILEB) + thr)
#define SG_ISSUE(buf, kt)                                                      \
    do {                                                                       \
        const unsigned go = (unsigned)((kt) * (BK * 2));                       \
        sg_async16(SG_DST(buf, 0),        offA + go,        baseA1);           \
        sg_async16(SG_DST(buf, 0) + 2560, offA + ROW1 + go, baseA1);           \
        sg_async16(SG_DST(buf, 1),        offA + go,        baseA2);           \
        sg_async16(SG_DST(buf, 1) + 2560, offA + ROW1 + go, baseA2);           \
        sg_async16(SG_DST(buf, 2),        offB + go,        baseB);            \
        sg_async16(SG_DST(buf, 2) + 2560, offB + ROW1 + go, baseB);            \
    } while (0)
#define SG_WAIT6()  asm volatile("s_wait_asynccnt 0x6" ::: "memory")
#define SG_WAIT0()  asm volatile("s_wait_asynccnt 0x0" ::: "memory")

    const int wid  = tid >> 5;
    const int lane = tid & 31;
    const int wm   = wid & 1;          // 2 wave rows (32 M each)
    const int wn   = wid >> 1;         // 2 wave cols (32 N each)
    const int lrow = lane & 15;
    const int lhi  = lane >> 4;
    const int ko   = lhi * 8;

    v8f acc[2][2][2] = {};             // [matrix(A1/A2)][m-tile][n-tile]

    SG_ISSUE(0, 0);
    SG_ISSUE(1, 1);

    // steady state: consume tile kk (buf kk%4), refill buf consumed 2 steps ago.
    // in-order ASYNC completion => wait(<=6) guarantees the consumed tile landed.
    for (int kk = 0; kk < KSTEPS - 4; kk += 4) {
        SG_WAIT6(); __syncthreads(); SG_ISSUE(2, kk + 2);
        sg_mma_step(lds[0][0], lds[0][1], lds[0][2], wm, wn, lrow, ko, acc);
        SG_WAIT6(); __syncthreads(); SG_ISSUE(3, kk + 3);
        sg_mma_step(lds[1][0], lds[1][1], lds[1][2], wm, wn, lrow, ko, acc);
        SG_WAIT6(); __syncthreads(); SG_ISSUE(0, kk + 4);
        sg_mma_step(lds[2][0], lds[2][1], lds[2][2], wm, wn, lrow, ko, acc);
        SG_WAIT6(); __syncthreads(); SG_ISSUE(1, kk + 5);
        sg_mma_step(lds[3][0], lds[3][1], lds[3][2], wm, wn, lrow, ko, acc);
    }
    // drain tail: tiles 124..127 live in bufs 0..3; issue the last two, then drain.
    SG_WAIT6(); __syncthreads(); SG_ISSUE(2, KSTEPS - 2);
    sg_mma_step(lds[0][0], lds[0][1], lds[0][2], wm, wn, lrow, ko, acc);
    SG_WAIT6(); __syncthreads(); SG_ISSUE(3, KSTEPS - 1);
    sg_mma_step(lds[1][0], lds[1][1], lds[1][2], wm, wn, lrow, ko, acc);
    SG_WAIT6(); __syncthreads();
    sg_mma_step(lds[2][0], lds[2][1], lds[2][2], wm, wn, lrow, ko, acc);
    SG_WAIT0(); __syncthreads();
    sg_mma_step(lds[3][0], lds[3][1], lds[3][2], wm, wn, lrow, ko, acc);

#undef SG_DST
#undef SG_ISSUE
#undef SG_WAIT6
#undef SG_WAIT0

    // ---------------- epilogue: each lane owns 2 target columns
    #pragma unroll
    for (int nt = 0; nt < 2; ++nt) {
        const int t = bn0 + wn * 32 + nt * 16 + lrow;
        if (t >= TT) continue;

        const float4 tb = tboxes[t];
        const float tx0 = tb.x - 0.5f * tb.z, ty0 = tb.y - 0.5f * tb.w;
        const float tx1 = tb.x + 0.5f * tb.z, ty1 = tb.y + 0.5f * tb.w;
        const float area_b = tb.z * tb.w;
        const float cs_t   = colsum_t[t];
        const int   lbl    = labels[t];

        #pragma unroll
        for (int mt = 0; mt < 2; ++mt) {
            #pragma unroll
            for (int i = 0; i < 8; ++i) {
                const int p = bm0 + wm * 32 + mt * 16 + lhi * 8 + i;
                const float g1 = acc[0][mt][nt][i];
                const float g2 = acc[1][mt][nt][i];

                const float4 pb = pboxes[p];
                const float cost_bbox = fabsf(pb.x - tb.x) + fabsf(pb.y - tb.y) +
                                        fabsf(pb.z - tb.z) + fabsf(pb.w - tb.w);
                const float ax0 = pb.x - 0.5f * pb.z, ay0 = pb.y - 0.5f * pb.w;
                const float ax1 = pb.x + 0.5f * pb.z, ay1 = pb.y + 0.5f * pb.w;
                const float area_a = pb.z * pb.w;
                const float iw = fmaxf(fminf(ax1, tx1) - fmaxf(ax0, tx0), 0.f);
                const float ih = fmaxf(fminf(ay1, ty1) - fmaxf(ay0, ty0), 0.f);
                const float inter = iw * ih;
                const float uni   = area_a + area_b - inter;
                const float iou   = inter / uni;
                const float ew = fmaxf(fmaxf(ax1, tx1) - fminf(ax0, tx0), 0.f);
                const float eh = fmaxf(fmaxf(ay1, ty1) - fminf(ay0, ty0), 0.f);
                const float enc = ew * eh;
                const float giou = iou - (enc - uni) / enc;

                const float cost_class = -probs[(size_t)p * NC + lbl];
                const float cost_mask  = -(g1 + sum_lognp[p]) * (1.0f / (float)KK);
                const float cost_dice  = 1.0f - (2.0f * g2 + EPSV) / (sum_sig[p] + cs_t + EPSV);

                out[(size_t)p * TT + t] = 5.0f * cost_bbox + 2.0f * (-giou) +
                                          2.0f * cost_class + 5.0f * cost_mask +
                                          5.0f * cost_dice;
            }
        }
    }
}

// ---------------------------------------------------------------- launcher
extern "C" void kernel_launch(void* const* d_in, const int* in_sizes, int n_in,
                              void* d_out, int out_size, void* d_ws, size_t ws_size,
                              hipStream_t stream) {
    (void)in_sizes; (void)n_in; (void)out_size; (void)ws_size;

    const float*  logits = (const float*)d_in[0];   // [16,300,81]
    const float4* pboxes = (const float4*)d_in[1];  // [16,300,4]
    const float*  pmasks = (const float*)d_in[2];   // [16,300,64,64]
    const float4* tboxes = (const float4*)d_in[3];  // [400,4]
    const float*  tmasks = (const float*)d_in[4];   // [400,128,128]
    const int*    labels = (const int*)d_in[5];     // [400]
    float*        out    = (float*)d_out;           // [16,300,400]

    char* ws = (char*)d_ws;
    size_t off = 0;
    _Float16* Amat = (_Float16*)(ws + off); off += (size_t)2 * PN * KK * sizeof(_Float16); // 78.6 MB
    _Float16* Bmat = (_Float16*)(ws + off); off += (size_t)TP * KK * sizeof(_Float16);     //  3.7 MB
    float* probs     = (float*)(ws + off); off += (size_t)PN * NC * sizeof(float);
    float* sum_lognp = (float*)(ws + off); off += (size_t)PN * sizeof(float);
    float* sum_sig   = (float*)(ws + off); off += (size_t)PN * sizeof(float);
    float* colsum_t  = (float*)(ws + off); off += (size_t)TP * sizeof(float);

    sg_softmax<<<PN / 8, 256, 0, stream>>>(logits, probs);
    sg_prep_pred<<<PN, 256, 0, stream>>>(pmasks, Amat, sum_lognp, sum_sig);
    sg_prep_tgt<<<TP, 256, 0, stream>>>(tmasks, Bmat, colsum_t);

    dim3 grid(TP / BN, PN / BM);  // (7, 75)
    sg_gemm_epilogue<<<grid, 128, 0, stream>>>(Amat, Bmat, pboxes, tboxes, probs,
                                               sum_lognp, sum_sig, colsum_t, labels, out);
}
